// LK_38792144617517
// MI455X (gfx1250) — matrix operand order repaired and checked by reading, not compiled, not run on hardware
//
#include <hip/hip_runtime.h>
#include <hip/hip_bf16.h>

typedef __attribute__((ext_vector_type(16))) _Float16 v16h;
typedef __attribute__((ext_vector_type(8)))  _Float16 v8h;
typedef __attribute__((ext_vector_type(8)))  float    v8f;
typedef __attribute__((vector_size(16)))     int      v4i;

// ---------------- problem constants (from reference) ----------------
#define BB     8
#define SS     16
#define NIMG   128           // B*S
#define CIN    3
#define HH     256
#define WW     256
#define HWPIX  65536         // 256*256
#define PFILT  68
#define KC     147           // 3*7*7
#define KPAD   160           // 5 chunks of 32
#define NPAD   80            // 5 tiles of 16
#define ASTR   168           // A-tile row stride in halfs (336B, 16B aligned, bank-spread)
#define PAIRS  120           // B*(S-1)
#define KLK    289           // 17*17 window
#define LKSTEPS 10

// =====================================================================
// Weight pack: det_w [68][3][7][7] f32 -> Wpack [KPAD][NPAD] f16, zero pad
// =====================================================================
__global__ void pack_weights(const float* __restrict__ detw,
                             _Float16* __restrict__ wpack) {
  int i = blockIdx.x * blockDim.x + threadIdx.x;
  if (i >= KPAD * NPAD) return;
  int k = i / NPAD;
  int n = i - k * NPAD;
  float v = 0.0f;
  if (k < KC && n < PFILT) v = detw[n * KC + k];   // k = c*49 + ky*7 + kx
  wpack[i] = (_Float16)v;
}

// =====================================================================
// Grayscale: mean over channels, float4 vectorized (stream 133 MB)
// =====================================================================
__global__ void to_gray(const float* __restrict__ in, float* __restrict__ gray) {
  size_t i = ((size_t)blockIdx.x * blockDim.x + threadIdx.x) * 4;
  if (i >= (size_t)NIMG * HWPIX) return;
  size_t img = i >> 16;
  size_t pix = i & 65535u;
  const float* b = in + img * (size_t)(CIN * HWPIX) + pix;
  float4 c0 = *(const float4*)(b);
  float4 c1 = *(const float4*)(b + HWPIX);
  float4 c2 = *(const float4*)(b + 2 * HWPIX);
  float4 o;
  o.x = (c0.x + c1.x + c2.x) * (1.0f / 3.0f);
  o.y = (c0.y + c1.y + c2.y) * (1.0f / 3.0f);
  o.z = (c0.z + c1.z + c2.z) * (1.0f / 3.0f);
  o.w = (c0.w + c1.w + c2.w) * (1.0f / 3.0f);
  *(float4*)(gray + i) = o;
}

// =====================================================================
// Conv (im2col GEMM via WMMA f16 16x16x32) fused with online softmax
// soft-argmax partials.  Block = 160 threads = 5 waves; wave w owns
// filter tile w (16 filters).  grid = (8 M-splits, 128 images).
// Double-buffered A tile in LDS: next tile's im2col gather overlaps
// the current tile's 5 WMMAs; one barrier per M-tile.
// =====================================================================
__launch_bounds__(160)
__global__ void conv_softargmax(const float* __restrict__ in,
                                const _Float16* __restrict__ wpack,
                                float* __restrict__ stats /* [128][80][8][4] */) {
  __shared__ __attribute__((aligned(16))) _Float16 Alds[2][16 * ASTR];
  __shared__ __attribute__((aligned(16))) _Float16 Blds[KPAD * NPAD];

  const int tid   = threadIdx.x;       // 0..159
  const int lane  = tid & 31;
  const int wave  = tid >> 5;          // N-tile 0..4
  const int split = blockIdx.x;        // 0..7
  const int img   = blockIdx.y;        // 0..127

  // --- stage B panel (25.6 KB) into LDS, async path when available ---
#if defined(__gfx1250__) && __has_builtin(__builtin_amdgcn_global_load_async_to_lds_b128)
  {
    _Float16* wp = const_cast<_Float16*>(wpack);
    for (int i = tid; i < (KPAD * NPAD) / 8; i += 160) {   // 1600 x 16B
      __builtin_amdgcn_global_load_async_to_lds_b128(
          (__attribute__((address_space(1))) v4i*)(wp + i * 8),
          (__attribute__((address_space(3))) v4i*)(Blds + i * 8), 0, 0);
    }
  }
#if __has_builtin(__builtin_amdgcn_s_wait_asynccnt)
  __builtin_amdgcn_s_wait_asynccnt(0);
#else
  asm volatile("s_wait_asynccnt 0" ::: "memory");
#endif
#else
  for (int i = tid; i < KPAD * NPAD; i += 160) Blds[i] = wpack[i];
#endif
  __syncthreads();

  // B fragments for this wave's filter tile, resident across all M-tiles.
  // this code's k order: lanes 0-15 hold K=0..15, lanes 16-31 K=16..31,
  // two K rows per VGPR (packed halfs), N = lane&15.
  v16h Bfrag[5];
  {
    const int n    = lane & 15;
    const int krow = (lane < 16) ? 0 : 16;
    for (int c = 0; c < 5; ++c)
      for (int v = 0; v < 8; ++v) {
        Bfrag[c][2 * v]     = Blds[(c * 32 + krow + 2 * v)     * NPAD + wave * 16 + n];
        Bfrag[c][2 * v + 1] = Blds[(c * 32 + krow + 2 * v + 1) * NPAD + wave * 16 + n];
      }
  }

  const float* imgbase = in + (size_t)img * (CIN * HWPIX);

  // im2col staging of one 16x160 A tile into buffer `buf`
  auto stageA = [&](int buf, int mtile) {
    const int mbase = (split * 32 + mtile) * 16;
    _Float16* A = &Alds[buf][0];
    for (int i = tid; i < 16 * KPAD; i += 160) {
      int mrow = i / KPAD;
      int k    = i - mrow * KPAD;
      float val = 0.0f;
      if (k < KC) {
        int c  = k / 49;
        int r  = k - c * 49;
        int ky = r / 7;
        int kx = r - ky * 7;
        int sp = mbase + mrow;
        int oy = sp >> 6, ox = sp & 63;
        int yy = oy * 4 + ky - 1;               // SAME pad: lo=1, hi=2
        int xx = ox * 4 + kx - 1;
        if ((unsigned)yy < (unsigned)HH && (unsigned)xx < (unsigned)WW)
          val = imgbase[c * HWPIX + yy * WW + xx];
      }
      A[mrow * ASTR + k] = (_Float16)val;
    }
  };

  // online softmax accumulators (this lane's filter = wave*16 + (lane&15))
  float mrun = -1.0e30f, srun = 0.0f, sxrun = 0.0f, syrun = 0.0f;

  stageA(0, 0);
  __syncthreads();

  const int m    = lane & 15;
  const int lo   = (lane < 16) ? 0 : 16;
  const int madd = (lane < 16) ? 0 : 8;        // D: VGPR r -> M=r or 8+r

  for (int mt = 0; mt < 32; ++mt) {
    const int buf   = mt & 1;
    const int mbase = (split * 32 + mt) * 16;
    const _Float16* A = &Alds[buf][0];

    // preload all 5 A fragments (10 x ds_load_b128, single wait)
    v16h Afrag[5];
    for (int c = 0; c < 5; ++c) {
      v8h alo = *(const v8h*)(A + m * ASTR + c * 32 + lo);
      v8h ahi = *(const v8h*)(A + m * ASTR + c * 32 + 8 + lo);
      for (int e = 0; e < 8; ++e) { Afrag[c][e] = alo[e]; Afrag[c][8 + e] = ahi[e]; }
    }

    // kick off next tile's global gather while WMMAs run
    if (mt + 1 < 32) stageA(buf ^ 1, mt + 1);

    v8f acc = {};
    for (int c = 0; c < 5; ++c)
      acc = __builtin_amdgcn_wmma_f32_16x16x32_f16(false, Afrag[c], false, Bfrag[c],
                                                   (short)0, acc, false, false);

    // online softmax / soft-argmax epilogue: tree max, single rescale,
    // 8 independent exps (good TRANS-pipe ILP)
    float tm = fmaxf(fmaxf(fmaxf(acc[0], acc[1]), fmaxf(acc[2], acc[3])),
                     fmaxf(fmaxf(acc[4], acc[5]), fmaxf(acc[6], acc[7])));
    float nm = fmaxf(mrun, tm);
    float e0 = __expf(mrun - nm);
    float sa = 0.0f, sxa = 0.0f, sya = 0.0f;
    for (int r = 0; r < 8; ++r) {
      int   sp = mbase + madd + r;
      float e1 = __expf(acc[r] - nm);
      sa  += e1;
      sxa += (float)(sp & 63) * e1;
      sya += (float)(sp >> 6) * e1;
    }
    srun  = srun  * e0 + sa;
    sxrun = sxrun * e0 + sxa;
    syrun = syrun * e0 + sya;
    mrun  = nm;

    __syncthreads();   // staging of buf^1 complete before next iteration reads it
  }

  // merge the two half-lanes holding the same filter (lane, lane+16)
  float m2  = __shfl_down(mrun,  16, 32);
  float s2  = __shfl_down(srun,  16, 32);
  float sx2 = __shfl_down(sxrun, 16, 32);
  float sy2 = __shfl_down(syrun, 16, 32);
  if (lane < 16) {
    float nm = fmaxf(mrun, m2);
    float e0 = __expf(mrun - nm), e1 = __expf(m2 - nm);
    float s  = srun  * e0 + s2  * e1;
    float sx = sxrun * e0 + sx2 * e1;
    float sy = syrun * e0 + sy2 * e1;
    int p = wave * 16 + lane;
    float* st = stats + (((size_t)img * NPAD + p) * 8 + split) * 4;
    st[0] = nm; st[1] = s; st[2] = sx; st[3] = sy;
  }
}

// =====================================================================
// Merge 8 partial softmax stats per (image, filter) -> locs [N][P][2]
// =====================================================================
__global__ void merge_stats(const float* __restrict__ stats,
                            float* __restrict__ locs) {
  int idx = blockIdx.x * blockDim.x + threadIdx.x;
  if (idx >= NIMG * PFILT) return;
  int img = idx / PFILT;
  int p   = idx - img * PFILT;
  const float* st = stats + (((size_t)img * NPAD + p) * 8) * 4;
  float m = -1.0e30f, s = 0.0f, sx = 0.0f, sy = 0.0f;
  for (int i = 0; i < 8; ++i) {
    float mi  = st[i * 4 + 0], si = st[i * 4 + 1];
    float sxi = st[i * 4 + 2], syi = st[i * 4 + 3];
    float nm  = fmaxf(m, mi);
    float e0  = __expf(m - nm), e1 = __expf(mi - nm);
    s = s * e0 + si * e1; sx = sx * e0 + sxi * e1; sy = sy * e0 + syi * e1;
    m = nm;
  }
  float inv = 1.0f / s;
  locs[(size_t)idx * 2 + 0] = sx * inv * 4.0f;   // * STRIDE
  locs[(size_t)idx * 2 + 1] = sy * inv * 4.0f;
}

// =====================================================================
// Lucas-Kanade: one 128-thread block (4 waves) per (pair, point).
// gray frames (33.5 MB) are L2-resident on MI455X (192 MB L2).
// Reductions: wave shfl_xor + 4-entry LDS combine (2 barriers each).
// =====================================================================
__device__ __forceinline__ float bilin(const float* __restrict__ img,
                                       float x, float y) {
  float xf = floorf(x), yf = floorf(y);
  float wx = x - xf, wy = y - yf;
  int x0 = (int)xf, y0 = (int)yf;
  x0 = x0 < 0 ? 0 : (x0 > WW - 1 ? WW - 1 : x0);
  y0 = y0 < 0 ? 0 : (y0 > HH - 1 ? HH - 1 : y0);
  int x1 = (x0 + 1 > WW - 1) ? WW - 1 : x0 + 1;
  int y1 = (y0 + 1 > HH - 1) ? HH - 1 : y0 + 1;
  float v00 = img[y0 * WW + x0], v01 = img[y0 * WW + x1];
  float v10 = img[y1 * WW + x0], v11 = img[y1 * WW + x1];
  return v00 * (1.0f - wx) * (1.0f - wy) + v01 * wx * (1.0f - wy)
       + v10 * (1.0f - wx) * wy         + v11 * wx * wy;
}

__device__ __forceinline__ void waveRed(float& v) {
  v += __shfl_xor(v, 16, 32);
  v += __shfl_xor(v,  8, 32);
  v += __shfl_xor(v,  4, 32);
  v += __shfl_xor(v,  2, 32);
  v += __shfl_xor(v,  1, 32);
}

__device__ __forceinline__ void blockRed3(float& a, float& b, float& c,
                                          float (*sh)[3]) {
  waveRed(a); waveRed(b); waveRed(c);
  int w = threadIdx.x >> 5, lane = threadIdx.x & 31;
  if (lane == 0) { sh[w][0] = a; sh[w][1] = b; sh[w][2] = c; }
  __syncthreads();
  a = sh[0][0] + sh[1][0] + sh[2][0] + sh[3][0];
  b = sh[0][1] + sh[1][1] + sh[2][1] + sh[3][1];
  c = sh[0][2] + sh[1][2] + sh[2][2] + sh[3][2];
  __syncthreads();
}

__device__ __forceinline__ void blockRed2(float& a, float& b, float (*sh)[3]) {
  waveRed(a); waveRed(b);
  int w = threadIdx.x >> 5, lane = threadIdx.x & 31;
  if (lane == 0) { sh[w][0] = a; sh[w][1] = b; }
  __syncthreads();
  a = sh[0][0] + sh[1][0] + sh[2][0] + sh[3][0];
  b = sh[0][1] + sh[1][1] + sh[2][1] + sh[3][1];
  __syncthreads();
}

__launch_bounds__(128)
__global__ void lk_track(const float* __restrict__ gray,
                         const float* __restrict__ ptsIn,
                         int inIsLocs, int inOff, int iOff, int jOff,
                         float* __restrict__ ptsOut) {
  const int blk  = blockIdx.x;            // pair*68 + p
  const int pair = blk / PFILT;
  const int p    = blk - pair * PFILT;
  const int b    = pair / (SS - 1);
  const int t    = pair - b * (SS - 1);
  const float* I = gray + (size_t)(b * SS + t + iOff) * HWPIX;
  const float* J = gray + (size_t)(b * SS + t + jOff) * HWPIX;
  size_t inIdx = inIsLocs ? ((size_t)(b * SS + t + inOff) * PFILT + p) * 2
                          : ((size_t)pair * PFILT + p) * 2;
  const float p0x = ptsIn[inIdx];
  const float p0y = ptsIn[inIdx + 1];

  const int tid = threadIdx.x;
  __shared__ float sh[4][3];

  float T[3], Ixv[3], Iyv[3], offx[3], offy[3];
  float a11 = 0.0f, a12 = 0.0f, a22 = 0.0f;
  for (int s = 0; s < 3; ++s) {
    int k = tid + s * 128;
    float ox = (float)(k % 17) - 8.0f;
    float oy = (float)(k / 17) - 8.0f;
    offx[s] = ox; offy[s] = oy;
    float t0 = 0.0f, ix = 0.0f, iy = 0.0f;
    if (k < KLK) {
      float px = p0x + ox, py = p0y + oy;
      if (s == 0) {   // warm the cache for the J window (global_prefetch_b8)
        int xi = (int)px; xi = xi < 0 ? 0 : (xi > WW - 1 ? WW - 1 : xi);
        int yi = (int)py; yi = yi < 0 ? 0 : (yi > HH - 1 ? HH - 1 : yi);
        __builtin_prefetch(J + yi * WW + xi, 0, 0);
      }
      t0 = bilin(I, px, py);
      ix = (bilin(I, px + 1.0f, py) - bilin(I, px - 1.0f, py)) * 0.5f;
      iy = (bilin(I, px, py + 1.0f) - bilin(I, px, py - 1.0f)) * 0.5f;
    }
    T[s] = t0; Ixv[s] = ix; Iyv[s] = iy;
    a11 += ix * ix; a12 += ix * iy; a22 += iy * iy;
  }
  blockRed3(a11, a12, a22, sh);
  a11 += 1e-6f;
  a22 += 1e-6f;
  float det = a11 * a22 - a12 * a12;
  float i11 =  a22 / det, i12 = -a12 / det, i22 = a11 / det;

  float ptx = p0x, pty = p0y;
  for (int step = 0; step < LKSTEPS; ++step) {
    float bx = 0.0f, by = 0.0f;
    for (int s = 0; s < 3; ++s) {
      int k = tid + s * 128;
      if (k < KLK) {
        float e = T[s] - bilin(J, ptx + offx[s], pty + offy[s]);
        bx += Ixv[s] * e;
        by += Iyv[s] * e;
      }
    }
    blockRed2(bx, by, sh);
    ptx += i11 * bx + i12 * by;
    pty += i12 * bx + i22 * by;
  }
  if (tid == 0) {
    ptsOut[((size_t)pair * PFILT + p) * 2 + 0] = ptx;
    ptsOut[((size_t)pair * PFILT + p) * 2 + 1] = pty;
  }
}

// =====================================================================
// Launch
// =====================================================================
extern "C" void kernel_launch(void* const* d_in, const int* in_sizes, int n_in,
                              void* d_out, int out_size, void* d_ws, size_t ws_size,
                              hipStream_t stream) {
  (void)in_sizes; (void)n_in; (void)out_size; (void)ws_size;
  const float* inputs = (const float*)d_in[0];
  const float* det_w  = (const float*)d_in[1];
  // d_in[2] (det_b): constant over the softmax axis -> cancels; unused.
  // d_in[3] (use_lk): fixed to 1 by the harness setup.

  char* ws = (char*)d_ws;
  float*    gray  = (float*)ws;                                     // 33,554,432 B
  _Float16* wpack = (_Float16*)(ws + (size_t)33554432);             //     25,600 B
  float*    stats = (float*)(ws + (size_t)33554432 + 25600);        //  1,310,720 B

  float* out      = (float*)d_out;
  float* locs     = out;                     // [8][16][68][2] = 17408
  float* nextPts  = out + 17408;             // [8][15][68][2] = 16320
  float* fbackPts = out + 17408 + 16320;
  float* backPts  = out + 17408 + 2 * 16320;

  pack_weights<<<(KPAD * NPAD + 255) / 256, 256, 0, stream>>>(det_w, wpack);
  to_gray<<<(NIMG * HWPIX / 4 + 255) / 256, 256, 0, stream>>>(inputs, gray);
  conv_softargmax<<<dim3(8, NIMG), 160, 0, stream>>>(inputs, wpack, stats);
  merge_stats<<<(NIMG * PFILT + 255) / 256, 256, 0, stream>>>(stats, locs);

  // pass 1: template I=frame t at locs[t]  -> nextPts
  lk_track<<<PAIRS * PFILT, 128, 0, stream>>>(gray, locs,    1, 0, 0, 1, nextPts);
  // pass 2: template J=frame t+1 at nextPts -> fbackPts (depends on pass 1)
  lk_track<<<PAIRS * PFILT, 128, 0, stream>>>(gray, nextPts, 0, 0, 1, 0, fbackPts);
  // pass 3: template J=frame t+1 at locs[t+1] -> backPts
  lk_track<<<PAIRS * PFILT, 128, 0, stream>>>(gray, locs,    1, 1, 1, 0, backPts);
}
